// MultiHeadSelfAttention_64802466562101
// MI455X (gfx1250) — matrix operand order repaired and checked by reading, not compiled, not run on hardware
//
#include <hip/hip_runtime.h>

#define DMODEL 1024
#define NHEADS 16
#define DHEAD  64
#define BSZ    2
#define TLEN   2048
#define MTOT   (BSZ * TLEN)     // 4096 tokens
#define NQKV   (3 * DMODEL)     // 3072

typedef __attribute__((ext_vector_type(16))) __bf16 v16bf;
typedef __attribute__((ext_vector_type(8)))  float  v8f;

union FragBF {
  v16bf v;
  uint4 q[2];
  unsigned short s[16];
};

__device__ __forceinline__ unsigned short f32_to_bf16(float f) {
  unsigned u = __builtin_bit_cast(unsigned, f);
  u += 0x7FFFu + ((u >> 16) & 1u);   // round-to-nearest-even
  return (unsigned short)(u >> 16);
}

__device__ __forceinline__ v8f wmma_bf16(const FragBF& a, const FragBF& b, v8f c) {
  return __builtin_amdgcn_wmma_f32_16x16x32_bf16(false, a.v, false, b.v,
                                                 (short)0, c, false, false);
}

__device__ __forceinline__ v8f vzero8() {
  v8f z = {0.f, 0.f, 0.f, 0.f, 0.f, 0.f, 0.f, 0.f};
  return z;
}

// ---------------- conversion kernels ----------------

__global__ __launch_bounds__(256)
void cvt_f32_bf16(const float* __restrict__ src, unsigned short* __restrict__ dst, int n) {
  int i = blockIdx.x * blockDim.x + threadIdx.x;
  if (i < n) dst[i] = f32_to_bf16(src[i]);
}

// src [rows][cols] f32 -> dst [cols][rows] bf16  (so GEMM B-fragments load contiguously)
__global__ __launch_bounds__(256)
void transpose_f32_bf16(const float* __restrict__ src, unsigned short* __restrict__ dst,
                        int rows, int cols) {
  int i = blockIdx.x * blockDim.x + threadIdx.x;
  if (i < rows * cols) {
    int r = i / cols, c = i - r * cols;
    dst[(size_t)c * rows + r] = f32_to_bf16(src[i]);
  }
}

// ---------------- QKV projection: [4096,1024] @ [1024,3072] + bias ----------------
// Each wave: 64x64 output tile (4x4 WMMA frags -> 16 wmma per 16 b128 loads per
// k-step, ~32 FLOP/B from L2). Stores scatter into Q [B,H,T,d], K [B,H,T,d],
// V^T [B,H,d,T] bf16 for the attention kernel.

__global__ __launch_bounds__(256)
void qkv_gemm(const unsigned short* __restrict__ xb,   // [4096][1024] bf16
              const unsigned short* __restrict__ wT,   // [3072][1024] bf16 (w_qkv^T)
              const float* __restrict__ bias,          // [3072]
              unsigned short* __restrict__ Qb,
              unsigned short* __restrict__ Kb,
              unsigned short* __restrict__ Vt) {
  const int lane = threadIdx.x & 31;
  const int wave = (blockIdx.x * blockDim.x + threadIdx.x) >> 5;
  const int wavesN = NQKV / 64;                 // 48
  const int m0 = (wave / wavesN) * 64;
  const int n0 = (wave % wavesN) * 64;
  const int q = lane & 15, h = lane >> 4;

  v8f acc[4][4];
  #pragma unroll
  for (int i = 0; i < 4; i++)
    #pragma unroll
    for (int j = 0; j < 4; j++) acc[i][j] = vzero8();

  const unsigned short* aRow[4];
  const unsigned short* bRow[4];
  #pragma unroll
  for (int i = 0; i < 4; i++) {
    aRow[i] = xb + (size_t)(m0 + i * 16 + q) * DMODEL;
    bRow[i] = wT + (size_t)(n0 + i * 16 + q) * DMODEL;
  }

  for (int kk = 0; kk < DMODEL; kk += 32) {
    FragBF a[4], b[4];
    #pragma unroll
    for (int i = 0; i < 4; i++) {
      // A frag (16x32 bf16): elems 0-7 at k = kk+8h, elems 8-15 at k = kk+16+8h
      a[i].q[0] = *reinterpret_cast<const uint4*>(aRow[i] + kk + 8 * h);
      a[i].q[1] = *reinterpret_cast<const uint4*>(aRow[i] + kk + 16 + 8 * h);
      // B frag (32x16 bf16): elems 0-7 at k = kk+8h, elems 8-15 at k = kk+16+8h (w^T row n)
      b[i].q[0] = *reinterpret_cast<const uint4*>(bRow[i] + kk + 8 * h);
      b[i].q[1] = *reinterpret_cast<const uint4*>(bRow[i] + kk + 16 + 8 * h);
    }
    #pragma unroll
    for (int i = 0; i < 4; i++)
      #pragma unroll
      for (int j = 0; j < 4; j++)
        acc[i][j] = wmma_bf16(a[i], b[j], acc[i][j]);
  }

  #pragma unroll
  for (int i = 0; i < 4; i++)
    #pragma unroll
    for (int j = 0; j < 4; j++)
      #pragma unroll
      for (int r = 0; r < 8; r++) {
        int m = m0 + i * 16 + r + 8 * h;          // token index in [0,4096)
        int n = n0 + j * 16 + q;                  // qkv column
        float val = acc[i][j][r] + bias[n];
        unsigned short bv = f32_to_bf16(val);
        int bb = m >> 11, t = m & (TLEN - 1);
        if (n < DMODEL) {
          int head = n >> 6, dd = n & 63;
          Qb[((size_t)(bb * NHEADS + head) * TLEN + t) * DHEAD + dd] = bv;
        } else if (n < 2 * DMODEL) {
          int c = n - DMODEL; int head = c >> 6, dd = c & 63;
          Kb[((size_t)(bb * NHEADS + head) * TLEN + t) * DHEAD + dd] = bv;
        } else {
          int c = n - 2 * DMODEL; int head = c >> 6, dd = c & 63;
          Vt[((size_t)(bb * NHEADS + head) * DHEAD + dd) * TLEN + t] = bv;
        }
      }
}

// ---------------- flash attention ----------------
// One wave per (b, head, 16-query tile). Computes S^T = K.Q^T and O^T = V^T.P^T so
// query sits on the lane axis: softmax = register reductions + one shfl_xor(16),
// and the C->B relayout for P.V needs only 16 cross-half shuffles. No LDS, no
// barriers (causal trip counts differ per wave). Scores never touch memory.

__global__ __launch_bounds__(256)
void attn_kernel(const unsigned short* __restrict__ Qb,
                 const unsigned short* __restrict__ Kb,
                 const unsigned short* __restrict__ Vt,
                 unsigned short* __restrict__ ybf) {   // [4096][1024] bf16
  const int lane = threadIdx.x & 31;
  const int wave = (blockIdx.x * blockDim.x + threadIdx.x) >> 5;
  const int qt = wave & 127;
  const int bh = wave >> 7;
  const int bb = bh >> 4, hh = bh & 15;
  const int q = lane & 15, h = lane >> 4;
  const int qBase = qt * 16;
  const int qIdx = qBase + q;

  const unsigned short* Qh = Qb + (size_t)bh * TLEN * DHEAD;
  const unsigned short* Kh = Kb + (size_t)bh * TLEN * DHEAD;
  const unsigned short* Vh = Vt + (size_t)bh * DHEAD * TLEN;

  // Q^T as B fragments (k = d), reused across all key blocks
  FragBF bq[2];
  #pragma unroll
  for (int c = 0; c < 2; c++) {
    const unsigned short* p = Qh + (size_t)qIdx * DHEAD + c * 32 + 8 * h;
    bq[c].q[0] = *reinterpret_cast<const uint4*>(p);
    bq[c].q[1] = *reinterpret_cast<const uint4*>(p + 16);
  }

  v8f acc[4];
  #pragma unroll
  for (int t = 0; t < 4; t++) acc[t] = vzero8();
  float m_st = -__builtin_inff();
  float l_st = 0.0f;

  const int kend = ((qBase + 16 + 31) >> 5) << 5;   // causal: keys <= qBase+15
  for (int kb = 0; kb < kend; kb += 32) {
    // S^T tiles: keys kb..kb+15 (s0) and kb+16..kb+31 (s1)
    v8f s0 = vzero8(), s1 = vzero8();
    #pragma unroll
    for (int c = 0; c < 2; c++) {
      FragBF aK;
      const unsigned short* p0 = Kh + (size_t)(kb + q) * DHEAD + c * 32 + 8 * h;
      aK.q[0] = *reinterpret_cast<const uint4*>(p0);
      aK.q[1] = *reinterpret_cast<const uint4*>(p0 + 16);
      s0 = wmma_bf16(aK, bq[c], s0);
      const unsigned short* p1 = Kh + (size_t)(kb + 16 + q) * DHEAD + c * 32 + 8 * h;
      aK.q[0] = *reinterpret_cast<const uint4*>(p1);
      aK.q[1] = *reinterpret_cast<const uint4*>(p1 + 16);
      s1 = wmma_bf16(aK, bq[c], s1);
    }

    // scale 1/sqrt(64), causal mask, online softmax (per-query = per-lane)
    const float NEG = -__builtin_inff();
    float mx = NEG;
    #pragma unroll
    for (int r = 0; r < 8; r++) {
      int k0 = kb + r + 8 * h;
      float v0 = (k0 <= qIdx) ? s0[r] * 0.125f : NEG;
      s0[r] = v0; mx = fmaxf(mx, v0);
      int k1 = kb + 16 + r + 8 * h;
      float v1 = (k1 <= qIdx) ? s1[r] * 0.125f : NEG;
      s1[r] = v1; mx = fmaxf(mx, v1);
    }
    mx = fmaxf(mx, __shfl_xor(mx, 16, 32));
    float mnew = fmaxf(m_st, mx);
    float corr = __expf(m_st - mnew);
    float sum = 0.0f;
    #pragma unroll
    for (int r = 0; r < 8; r++) {
      float e0 = __expf(s0[r] - mnew); s0[r] = e0; sum += e0;
      float e1 = __expf(s1[r] - mnew); s1[r] = e1; sum += e1;
    }
    sum += __shfl_xor(sum, 16, 32);
    l_st = l_st * corr + sum;
    m_st = mnew;
    #pragma unroll
    for (int t = 0; t < 4; t++)
      #pragma unroll
      for (int r = 0; r < 8; r++) acc[t][r] *= corr;

    // Build P^T B-fragment (32 keys x 16 queries) from the two C-layout tiles:
    // only a half-swap (shfl_xor 16) is needed; query stays on the lane axis.
    FragBF pb;
    #pragma unroll
    for (int r = 0; r < 8; r++) {
      float o0 = __shfl_xor(s0[r], 16, 32);
      float o1 = __shfl_xor(s1[r], 16, 32);
      pb.s[r]     = f32_to_bf16(h == 0 ? s0[r] : o1);
      pb.s[r + 8] = f32_to_bf16(h == 0 ? o0 : s1[r]);
    }

    // O^T += V^T x P^T  (4 d-tiles of 16)
    #pragma unroll
    for (int t = 0; t < 4; t++) {
      FragBF aV;
      const unsigned short* p = Vh + (size_t)(t * 16 + q) * TLEN + kb + 16 * h;
      aV.q[0] = *reinterpret_cast<const uint4*>(p);
      aV.q[1] = *reinterpret_cast<const uint4*>(p + 8);
      acc[t] = wmma_bf16(aV, pb, acc[t]);
    }
  }

  float rinv = 1.0f / l_st;
  int row = bb * TLEN + qIdx;
  #pragma unroll
  for (int t = 0; t < 4; t++)
    #pragma unroll
    for (int r = 0; r < 8; r++) {
      int col = hh * DHEAD + t * 16 + r + 8 * h;
      ybf[(size_t)row * DMODEL + col] = f32_to_bf16(acc[t][r] * rinv);
    }
}

// ---------------- output projection: [4096,1024] @ [1024,1024] + bias -> f32 ----------------
// 64x64 per-wave register tile, same structure as qkv_gemm.

__global__ __launch_bounds__(256)
void out_gemm(const unsigned short* __restrict__ yb,   // [4096][1024] bf16
              const unsigned short* __restrict__ wT,   // [1024][1024] bf16 (w_out^T)
              const float* __restrict__ bias,          // [1024]
              float* __restrict__ out) {               // [4096][1024] f32
  const int lane = threadIdx.x & 31;
  const int wave = (blockIdx.x * blockDim.x + threadIdx.x) >> 5;
  const int wavesN = DMODEL / 64;                 // 16
  const int m0 = (wave / wavesN) * 64;
  const int n0 = (wave % wavesN) * 64;
  const int q = lane & 15, h = lane >> 4;

  v8f acc[4][4];
  #pragma unroll
  for (int i = 0; i < 4; i++)
    #pragma unroll
    for (int j = 0; j < 4; j++) acc[i][j] = vzero8();

  const unsigned short* aRow[4];
  const unsigned short* bRow[4];
  #pragma unroll
  for (int i = 0; i < 4; i++) {
    aRow[i] = yb + (size_t)(m0 + i * 16 + q) * DMODEL;
    bRow[i] = wT + (size_t)(n0 + i * 16 + q) * DMODEL;
  }

  for (int kk = 0; kk < DMODEL; kk += 32) {
    FragBF a[4], b[4];
    #pragma unroll
    for (int i = 0; i < 4; i++) {
      a[i].q[0] = *reinterpret_cast<const uint4*>(aRow[i] + kk + 8 * h);
      a[i].q[1] = *reinterpret_cast<const uint4*>(aRow[i] + kk + 16 + 8 * h);
      b[i].q[0] = *reinterpret_cast<const uint4*>(bRow[i] + kk + 8 * h);
      b[i].q[1] = *reinterpret_cast<const uint4*>(bRow[i] + kk + 16 + 8 * h);
    }
    #pragma unroll
    for (int i = 0; i < 4; i++)
      #pragma unroll
      for (int j = 0; j < 4; j++)
        acc[i][j] = wmma_bf16(a[i], b[j], acc[i][j]);
  }

  #pragma unroll
  for (int i = 0; i < 4; i++)
    #pragma unroll
    for (int j = 0; j < 4; j++)
      #pragma unroll
      for (int r = 0; r < 8; r++) {
        int m = m0 + i * 16 + r + 8 * h;
        int n = n0 + j * 16 + q;
        out[(size_t)m * DMODEL + n] = acc[i][j][r] + bias[n];
      }
}

// ---------------- launch ----------------

extern "C" void kernel_launch(void* const* d_in, const int* in_sizes, int n_in,
                              void* d_out, int out_size, void* d_ws, size_t ws_size,
                              hipStream_t stream) {
  const float* x     = (const float*)d_in[0];
  const float* w_qkv = (const float*)d_in[1];
  const float* b_qkv = (const float*)d_in[2];
  const float* w_out = (const float*)d_in[3];
  const float* b_out = (const float*)d_in[4];
  // d_in[5] = attn_mask: causal, handled analytically in attn_kernel

  char* ws = (char*)d_ws;
  size_t off = 0;
  auto alloc = [&](size_t bytes) -> void* {
    void* p = ws + off;
    off = (off + bytes + 255) & ~(size_t)255;
    return p;
  };
  unsigned short* xb     = (unsigned short*)alloc((size_t)MTOT * DMODEL * 2);
  unsigned short* wqkvT  = (unsigned short*)alloc((size_t)NQKV * DMODEL * 2);
  unsigned short* woutT  = (unsigned short*)alloc((size_t)DMODEL * DMODEL * 2);
  unsigned short* Qb     = (unsigned short*)alloc((size_t)BSZ * NHEADS * TLEN * DHEAD * 2);
  unsigned short* Kb     = (unsigned short*)alloc((size_t)BSZ * NHEADS * TLEN * DHEAD * 2);
  unsigned short* Vt     = (unsigned short*)alloc((size_t)BSZ * NHEADS * DHEAD * TLEN * 2);
  unsigned short* ybf    = (unsigned short*)alloc((size_t)MTOT * DMODEL * 2);

  // 1) precision conversion / weight transposes
  {
    int n = MTOT * DMODEL;                       // 4,194,304
    cvt_f32_bf16<<<n / 256, 256, 0, stream>>>(x, xb, n);
  }
  {
    int n = DMODEL * NQKV;                       // 3,145,728
    transpose_f32_bf16<<<n / 256, 256, 0, stream>>>(w_qkv, wqkvT, DMODEL, NQKV);
  }
  {
    int n = DMODEL * DMODEL;                     // 1,048,576
    transpose_f32_bf16<<<n / 256, 256, 0, stream>>>(w_out, woutT, DMODEL, DMODEL);
  }

  // 2) QKV projection: (4096/64)*(3072/64) = 3072 waves, 8 waves/block
  qkv_gemm<<<3072 / 8, 256, 0, stream>>>(xb, wqkvT, b_qkv, Qb, Kb, Vt);

  // 3) flash attention: B*H*(T/16) = 4096 waves
  attn_kernel<<<4096 / 8, 256, 0, stream>>>(Qb, Kb, Vt, ybf);

  // 4) output projection: (4096/64)*(1024/64) = 1024 waves
  out_gemm<<<1024 / 8, 256, 0, stream>>>(ybf, woutT, b_out, (float*)d_out);
}
